// GuidedFilterLayer_76699525972626
// MI455X (gfx1250) — matrix operand order repaired and checked by reading, not compiled, not run on hardware
//
#include <hip/hip_runtime.h>
#include <hip/hip_bf16.h>

typedef __attribute__((ext_vector_type(16))) __bf16 v16bf;
typedef __attribute__((ext_vector_type(8)))  float  v8f;

#define B_  4
#define C_  256
#define H_  256
#define W_  256
#define R_  64
#define HL  64
#define WL  64
#define NPIX_LR 4096     // 64*64
#define NPIX_HR 65536    // 256*256

__device__ __forceinline__ unsigned short f2bf(float f) {
  union { float f; unsigned u; } v; v.f = f;
  unsigned r = v.u + 0x7FFFu + ((v.u >> 16) & 1u);   // round-to-nearest-even bf16
  return (unsigned short)(r >> 16);
}

struct U4 { unsigned x, y, z, w; };
union Frag { U4 q[2]; unsigned short u[16]; v16bf v; };

// ---------------- kernel 0: weights -> bf16 (L2 resident) ----------------
__global__ void k0_cvt_weights(const float* __restrict__ wrg, const float* __restrict__ wrt,
                               const float* __restrict__ wua, const float* __restrict__ wub,
                               unsigned short* __restrict__ out) {
  int i = blockIdx.x * blockDim.x + threadIdx.x;   // 0..65535
  int seg = i >> 14, off = i & 16383;
  const float* src = (seg == 0) ? wrg : (seg == 1) ? wrt : (seg == 2) ? wua : wub;
  out[i] = f2bf(src[off]);
}

// ---- kernel 1: exact 4x bilinear downsample fused with C->R projection ----
// grid: B * (4096/16) = 1024 blocks, 256 threads (8 waves)
#define LDS1_STR 264   // 256 + 8 halves pad: 528B row stride -> conflict-free b128 reads
__global__ void __launch_bounds__(256)
k1_down_proj(const float* __restrict__ fenc, const float* __restrict__ fdec,
             const unsigned short* __restrict__ wbf,
             float* __restrict__ fg, float* __restrict__ ft) {
  __shared__ unsigned short ldsE[16 * LDS1_STR];
  __shared__ unsigned short ldsD[16 * LDS1_STR];
  const int t = threadIdx.x;
  const int tile = blockIdx.x;
  const int b = tile >> 8;
  const int pixbase = (tile & 255) * 16;

  // stage 1: build 16px x 256c bf16 tiles of downsampled F_enc/F_dec.
  // src = (i+0.5)*4 - 0.5 = 4i+1.5  ->  avg of cols/rows {4i+1, 4i+2}, weight 0.25
  {
    const int pxl = t & 15;
    const int c0  = t >> 4;
    const int pg  = pixbase + pxl;
    const int y = pg >> 6, x = pg & 63;
    const long base0 = (long)(4 * y + 1) * W_ + (4 * x + 1);
    const float* eb = fenc + (long)b * C_ * NPIX_HR;
    const float* db = fdec + (long)b * C_ * NPIX_HR;
#pragma unroll
    for (int k = 0; k < 16; ++k) {
      int c = c0 + k * 16;
      const float* pE = eb + (long)c * NPIX_HR + base0;
      const float* pD = db + (long)c * NPIX_HR + base0;
      float e = 0.25f * (pE[0] + pE[1] + pE[W_] + pE[W_ + 1]);
      float d = 0.25f * (pD[0] + pD[1] + pD[W_] + pD[W_ + 1]);
      ldsE[pxl * LDS1_STR + c] = f2bf(e);
      ldsD[pxl * LDS1_STR + c] = f2bf(d);
    }
  }
  __syncthreads();

  // stage 2: wave w -> (mat = w>>2 : fg/ft, rtile = w&3). D(16r x 16px) = W(16x32) * X(32x16)
  const int wv = t >> 5, lane = t & 31;
  const int mat = wv >> 2, rtile = wv & 3;
  const unsigned short* wsel = wbf + mat * (R_ * C_);
  const unsigned short* lds  = mat ? ldsD : ldsE;
  const int m = lane & 15, hi = lane >> 4;          // A: M=m, khalf=hi*8 ; B/D: N=m
  const unsigned short* arow = wsel + (rtile * 16 + m) * C_;
  v8f acc = {};
#pragma unroll
  for (int ks = 0; ks < 8; ++ks) {
    const int kb = ks * 32 + hi * 8;
    Frag a, bf;
    a.q[0] = *(const U4*)(arow + kb);               // K = kb..kb+7
    a.q[1] = *(const U4*)(arow + kb + 16);          // K = kb+16..kb+23
    const unsigned short* bp = lds + m * LDS1_STR + ks * 32 + hi * 8;
    bf.q[0] = *(const U4*)(bp);
    bf.q[1] = *(const U4*)(bp + 16);
    acc = __builtin_amdgcn_wmma_f32_16x16x32_bf16(false, a.v, false, bf.v,
                                                  (short)0, acc, false, false);
  }
  float* out = mat ? ft : fg;
  const int rbase = rtile * 16 + hi * 8;
#pragma unroll
  for (int v = 0; v < 8; ++v)
    out[((long)b * R_ + rbase + v) * NPIX_LR + pixbase + m] = acc[v];
}

// ---------------- kernel 2: 3x3 box stats -> a,b (per (b,r) plane) ----------------
__global__ void __launch_bounds__(256)
k2_stats(const float* __restrict__ fg, const float* __restrict__ ft,
         const float* __restrict__ epsp,
         float* __restrict__ a, float* __restrict__ bb) {
  __shared__ float sg[NPIX_LR];
  __shared__ float st[NPIX_LR];
  const int plane = blockIdx.x;
  const int t = threadIdx.x;
  const float* gp = fg + (long)plane * NPIX_LR;
  const float* tp = ft + (long)plane * NPIX_LR;
#pragma unroll
  for (int i = 0; i < 16; ++i) { int p = t + i * 256; sg[p] = gp[p]; st[p] = tp[p]; }
  __syncthreads();
  const float eps = *epsp;
  const float inv9 = 1.0f / 9.0f;
#pragma unroll 4
  for (int i = 0; i < 16; ++i) {
    int p = t + i * 256;
    int y = p >> 6, x = p & 63;
    float Sg = 0.f, St = 0.f, Sgg = 0.f, Sgt = 0.f;
    for (int dy = -1; dy <= 1; ++dy) {
      int yy = y + dy; if (yy < 0 || yy >= HL) continue;
      for (int dx = -1; dx <= 1; ++dx) {
        int xx = x + dx; if (xx < 0 || xx >= WL) continue;
        float g = sg[yy * WL + xx], tt = st[yy * WL + xx];
        Sg += g; St += tt; Sgg += g * g; Sgt += g * tt;
      }
    }
    float mg = Sg * inv9, mt = St * inv9;
    float av = (Sgt * inv9 - mg * mt) / (Sgg * inv9 - mg * mg + eps);
    a [(long)plane * NPIX_LR + p] = av;
    bb[(long)plane * NPIX_LR + p] = mt - av * mg;
  }
}

// ---- kernel 3: fused bilinear upsample + R->C projection + residual epilogue ----
// grid: B*H*W/16 = 16384 blocks, 512 threads (16 waves = 16 o-tiles)
#define LDS3_STR 72    // 64 + 8 halves pad -> conflict-free b128 reads
__global__ void __launch_bounds__(512)
k3_up_out(const float* __restrict__ fenc, const float* __restrict__ fdec,
          const unsigned short* __restrict__ wbf,
          const float* __restrict__ a, const float* __restrict__ bb,
          float* __restrict__ out) {
  __shared__ unsigned short ldsA[16 * LDS3_STR];
  __shared__ unsigned short ldsB[16 * LDS3_STR];
  const int t = threadIdx.x;
  const long pixbase = (long)blockIdx.x * 16;
  const int b = (int)(pixbase >> 16);
  const int rem = (int)(pixbase & 65535);   // h*256 + w0, 16 contiguous w, same h
  const int h = rem >> 8, w0 = rem & 255;

  // y interpolation is uniform across the block
  float sy = 0.25f * h - 0.375f; sy = sy < 0.f ? 0.f : sy;
  const int y0 = (int)sy;
  const int y1 = (y0 + 1 > 63) ? 63 : y0 + 1;
  const float fyw = sy - (float)y0;

  // warm GL2 / L0 for the big streaming tile (global_prefetch_b8)
  {
    const int o = t & 255;
    const float* pf = (t < 256) ? fenc : fdec;
    __builtin_prefetch(pf + (((long)b * C_ + o) << 16) + rem, 0, 1);
  }

  // stage 1: bilinear-upsample a/b into 16px x 64r bf16 tiles
  {
    const int halfsel = t >> 8;                 // 0 -> a, 1 -> b
    const float* src = halfsel ? bb : a;
    unsigned short* dst = halfsel ? ldsB : ldsA;
#pragma unroll
    for (int it = 0; it < 4; ++it) {
      const int idx = (t & 255) + it * 256;     // 0..1023 over (r,px)
      const int px = idx & 15;
      const int r  = idx >> 4;
      float sx = 0.25f * (float)(w0 + px) - 0.375f; sx = sx < 0.f ? 0.f : sx;
      const int x0 = (int)sx;
      const int x1 = (x0 + 1 > 63) ? 63 : x0 + 1;
      const float fxw = sx - (float)x0;
      const float* pl = src + ((long)b * R_ + r) * NPIX_LR;
      float v00 = pl[y0 * WL + x0], v01 = pl[y0 * WL + x1];
      float v10 = pl[y1 * WL + x0], v11 = pl[y1 * WL + x1];
      float vi = (1.f - fyw) * ((1.f - fxw) * v00 + fxw * v01)
               +        fyw  * ((1.f - fxw) * v10 + fxw * v11);
      dst[px * LDS3_STR + r] = f2bf(vi);
    }
  }
  __syncthreads();

  // stage 2: per wave, D(16o x 16px) = Wua(16x64)*Aup(64x16) and Wub*Bup (K=64 -> 2 steps)
  const int wv = t >> 5, lane = t & 31;
  const int m = lane & 15, hi = lane >> 4;
  const unsigned short* aro = wbf + 2 * (R_ * C_) + (wv * 16 + m) * R_;  // w_ua [C,R]
  const unsigned short* bro = wbf + 3 * (R_ * C_) + (wv * 16 + m) * R_;  // w_ub [C,R]
  v8f accA = {}, accB = {};
#pragma unroll
  for (int ks = 0; ks < 2; ++ks) {
    const int kb = ks * 32 + hi * 8;
    Frag fa, fb, xa, xb;
    fa.q[0] = *(const U4*)(aro + kb);      fa.q[1] = *(const U4*)(aro + kb + 16);
    fb.q[0] = *(const U4*)(bro + kb);      fb.q[1] = *(const U4*)(bro + kb + 16);
    const unsigned short* pA = ldsA + m * LDS3_STR + ks * 32 + hi * 8;
    const unsigned short* pB = ldsB + m * LDS3_STR + ks * 32 + hi * 8;
    xa.q[0] = *(const U4*)pA;  xa.q[1] = *(const U4*)(pA + 16);
    xb.q[0] = *(const U4*)pB;  xb.q[1] = *(const U4*)(pB + 16);
    accA = __builtin_amdgcn_wmma_f32_16x16x32_bf16(false, fa.v, false, xa.v,
                                                   (short)0, accA, false, false);
    accB = __builtin_amdgcn_wmma_f32_16x16x32_bf16(false, fb.v, false, xb.v,
                                                   (short)0, accB, false, false);
  }

  // epilogue: out = F_dec + coefA*F_enc + coefB ; per-VGPR stores are 64B contiguous
  const int obase = wv * 16 + hi * 8;
#pragma unroll
  for (int v = 0; v < 8; ++v) {
    const long gi = (((long)b * C_ + (obase + v)) << 16) + rem + m;
    out[gi] = fdec[gi] + accA[v] * fenc[gi] + accB[v];
  }
}

extern "C" void kernel_launch(void* const* d_in, const int* in_sizes, int n_in,
                              void* d_out, int out_size, void* d_ws, size_t ws_size,
                              hipStream_t stream) {
  const float* fenc = (const float*)d_in[0];
  const float* fdec = (const float*)d_in[1];
  const float* wrg  = (const float*)d_in[2];
  const float* wrt  = (const float*)d_in[3];
  const float* wua  = (const float*)d_in[4];
  const float* wub  = (const float*)d_in[5];
  const float* eps  = (const float*)d_in[6];
  float* out = (float*)d_out;

  char* ws = (char*)d_ws;
  unsigned short* wbf = (unsigned short*)ws;                 // 4*16384 bf16 = 128KB
  float* fg = (float*)(ws + 131072);                         // [B,R,64,64] f32, 4MB each
  float* ft = fg + (long)B_ * R_ * NPIX_LR;
  float* aa = ft + (long)B_ * R_ * NPIX_LR;
  float* bbuf = aa + (long)B_ * R_ * NPIX_LR;

  k0_cvt_weights<<<256, 256, 0, stream>>>(wrg, wrt, wua, wub, wbf);
  k1_down_proj <<<1024, 256, 0, stream>>>(fenc, fdec, wbf, fg, ft);
  k2_stats     <<<256, 256, 0, stream>>>(fg, ft, eps, aa, bbuf);
  k3_up_out    <<<16384, 512, 0, stream>>>(fenc, fdec, wbf, aa, bbuf, out);
}
